// EncoderLayer_30803505447237
// MI455X (gfx1250) — compile-verified, matrix operand order checked by reading, not run on hardware
//
#include <hip/hip_runtime.h>
#include <hip/hip_bf16.h>

// ---------------------------------------------------------------------------
// MI455X (gfx1250) encoder layer. wave32, WMMA 16x16x32 f16 -> f32.
// GEMM: double-buffered async global->LDS pipeline (ASYNCcnt) + WMMA.
// ---------------------------------------------------------------------------

typedef __attribute__((ext_vector_type(16))) _Float16 v16h;
typedef __attribute__((ext_vector_type(8)))  _Float16 v8h;
typedef __attribute__((ext_vector_type(8)))  float    v8f;

constexpr int B   = 2;
constexpr int S   = 2048;
constexpr int D   = 1024;
constexpr int H   = 16;
constexpr int DK  = 64;
constexpr int DFF = 4096;
constexpr int M   = B * S;          // 4096 rows of tokens
constexpr float EPS = 1e-5f;

// ---------------------------------------------------------------------------
// WMMA helper: D = A(16x32 f16) * B(32x16 f16) + C(f32)
// ---------------------------------------------------------------------------
static __device__ __forceinline__ v8f wmma_f16(v16h a, v16h b, v8f c) {
    return __builtin_amdgcn_wmma_f32_16x16x32_f16(
        /*neg_a=*/false, a, /*neg_b=*/false, b,
        /*c_mod=*/(short)0, c, /*reuse_a=*/false, /*reuse_b=*/false);
}

// A/B fragment load (mirrored layout). base = row 0 of a 16-row tile whose
// 32 K-halves start at base; stride in halves. Per ISA 16-bit A layout:
// lanes 0-15 hold K 0..7 & 16..23, lanes 16-31 hold K 8..15 & 24..31.
static __device__ __forceinline__ v16h frag16(const _Float16* base, int stride, int lane) {
    const _Float16* p = base + (size_t)(lane & 15) * stride;
    const int kb = (lane & 16) ? 8 : 0;
    v8h lo = *(const v8h*)(p + kb);
    v8h hi = *(const v8h*)(p + kb + 16);
    return __builtin_shufflevector(lo, hi, 0,1,2,3,4,5,6,7,8,9,10,11,12,13,14,15);
}

// CDNA5 async copy: 16 bytes global -> LDS, no VGPR data path (ASYNCcnt).
static __device__ __forceinline__ void async_b128(const void* gaddr, void* lds) {
    const unsigned           loff = (unsigned)(uintptr_t)lds;   // LDS offset = addr[31:0]
    const unsigned long long ga   = (unsigned long long)(uintptr_t)gaddr;
    asm volatile("global_load_async_to_lds_b128 %0, %1, off"
                 :: "v"(loff), "v"(ga) : "memory");
}
static __device__ __forceinline__ void async_wait_all() {
    asm volatile("s_wait_asynccnt 0x0" ::: "memory");
}

// ---------------------------------------------------------------------------
// Conversion / layout kernels
// ---------------------------------------------------------------------------
__global__ __launch_bounds__(256) void cvt_f32_to_f16(const float* __restrict__ in,
                                                      _Float16* __restrict__ out, long n) {
    long i = (long)blockIdx.x * 256 + threadIdx.x;
    if (i < n) out[i] = (_Float16)in[i];
}

// W[K,N] (f32, row-major) -> WT[N,K] (f16, row-major)
__global__ __launch_bounds__(256) void transpose_cvt(const float* __restrict__ W,
                                                     _Float16* __restrict__ WT,
                                                     int K, int N) {
    long o = (long)blockIdx.x * 256 + threadIdx.x;
    if (o >= (long)K * N) return;
    int n = (int)(o / K);
    int k = (int)(o % K);
    WT[o] = (_Float16)W[(size_t)k * N + n];
}

// vh [b*S+s, h*DK+dk] (f16) -> vT [((b*H+h)*DK+dk)*S + s] (f16)
__global__ __launch_bounds__(256) void vtrans_kernel(const _Float16* __restrict__ vh,
                                                     _Float16* __restrict__ vT) {
    size_t o = (size_t)blockIdx.x * 256 + threadIdx.x;   // total M*D
    int s  = (int)(o & (S - 1));
    size_t t = o >> 11;            // /S
    int dk = (int)(t & (DK - 1));
    int bh = (int)(t >> 6);
    int b  = bh >> 4;
    int h  = bh & (H - 1);
    vT[o] = vh[((size_t)(b * S + s)) * D + h * DK + dk];
}

// ---------------------------------------------------------------------------
// Tiled WMMA GEMM:  C[M,N] = A[M,K](f16) * Bt[N,K]^T(f16) + bias[N]
// Block 128x128, 256 threads = 8 waves in 4(M) x 2(N); each wave 32x64.
// Double-buffered async-DMA pipeline: tile i+1 streams into LDS while the
// WMMAs consume tile i. One barrier per K step.
// ---------------------------------------------------------------------------
template <typename OutT, bool RELU>
__global__ __launch_bounds__(256)
void gemm_wmma(const _Float16* __restrict__ A, const _Float16* __restrict__ Bt,
               const float* __restrict__ bias, OutT* __restrict__ C,
               int Mdim, int Ndim, int Kdim) {
    constexpr int LDT = 40;                 // padded halves per LDS row (80B)
    constexpr int TSZ = 128 * LDT;          // halves per tile buffer
    __shared__ __align__(16) _Float16 As[2 * TSZ];
    __shared__ __align__(16) _Float16 Bs[2 * TSZ];

    const int tid  = threadIdx.x;
    const int lane = tid & 31;
    const int wid  = tid >> 5;
    const int wm   = wid >> 1;              // 0..3
    const int wn   = wid & 1;               // 0..1
    const int mBase = blockIdx.y * 128;
    const int nBase = blockIdx.x * 128;
    (void)Mdim;

    auto stage = [&](int bufsel, int kt) {
#pragma unroll
        for (int i = tid; i < 512; i += 256) {
            const int row = i >> 2;
            const int c8  = (i & 3) * 8;
            async_b128(&A[(size_t)(mBase + row) * Kdim + kt + c8],
                       &As[bufsel * TSZ + row * LDT + c8]);
            async_b128(&Bt[(size_t)(nBase + row) * Kdim + kt + c8],
                       &Bs[bufsel * TSZ + row * LDT + c8]);
        }
    };

    v8f zero = {};
    v8f acc[2][4];
#pragma unroll
    for (int mi = 0; mi < 2; ++mi)
#pragma unroll
        for (int ni = 0; ni < 4; ++ni) acc[mi][ni] = zero;

    stage(0, 0);                            // prologue: tile 0 in flight
    int buf = 0;
    for (int kt = 0; kt < Kdim; kt += 32, buf ^= 1) {
        async_wait_all();                   // tile kt has landed in LDS
        __syncthreads();                    // visible to all waves; all waves
                                            // are done reading buffer buf^1
        if (kt + 32 < Kdim) {
            stage(buf ^ 1, kt + 32);        // DMA next tile under the WMMAs
            if (kt + 64 < Kdim) {           // L2 prefetch two tiles ahead
                const int row = tid >> 1, c16 = (tid & 1) * 16;
                __builtin_prefetch(&A[(size_t)(mBase + row) * Kdim + kt + 64 + c16], 0, 1);
                __builtin_prefetch(&Bt[(size_t)(nBase + row) * Kdim + kt + 64 + c16], 0, 1);
            }
        }

        const _Float16* Ab = &As[buf * TSZ];
        const _Float16* Bb = &Bs[buf * TSZ];
        v16h af[2], bf[4];
#pragma unroll
        for (int mi = 0; mi < 2; ++mi)
            af[mi] = frag16(&Ab[(wm * 32 + mi * 16) * LDT], LDT, lane);
#pragma unroll
        for (int ni = 0; ni < 4; ++ni)
            bf[ni] = frag16(&Bb[(wn * 64 + ni * 16) * LDT], LDT, lane);

#pragma unroll
        for (int mi = 0; mi < 2; ++mi)
#pragma unroll
            for (int ni = 0; ni < 4; ++ni)
                acc[mi][ni] = wmma_f16(af[mi], bf[ni], acc[mi][ni]);
    }

    // Epilogue. C layout: VGPR i -> row i (lanes 0-15) / 8+i (lanes 16-31),
    // col = lane&15.
    const int hi8  = (lane & 16) ? 8 : 0;
    const int l16  = lane & 15;
#pragma unroll
    for (int mi = 0; mi < 2; ++mi) {
#pragma unroll
        for (int ni = 0; ni < 4; ++ni) {
            const int col = nBase + wn * 64 + ni * 16 + l16;
            const float bv = bias ? bias[col] : 0.0f;
#pragma unroll
            for (int i = 0; i < 8; ++i) {
                const int row = mBase + wm * 32 + mi * 16 + i + hi8;
                float v = acc[mi][ni][i] + bv;
                if (RELU) v = v > 0.0f ? v : 0.0f;
                C[(size_t)row * Ndim + col] = (OutT)v;
            }
        }
    }
}

// ---------------------------------------------------------------------------
// Flash attention. One block = 4 waves = 64 queries of one (b,h).
// Computes S^T = K * Q^T tiles so softmax is lane-local; the S^T accumulator
// layout coincides with the A-fragment layout for P*V (no cross-lane moves).
// ---------------------------------------------------------------------------
__global__ __launch_bounds__(128)
void attn_wmma(const _Float16* __restrict__ qh, const _Float16* __restrict__ kh,
               const _Float16* __restrict__ vT, _Float16* __restrict__ oh) {
    const int qtile = blockIdx.x;            // 0 .. S/64-1
    const int bh    = blockIdx.y;            // 0 .. B*H-1
    const int b     = bh / H;
    const int h     = bh % H;
    const int tid   = threadIdx.x;
    const int lane  = tid & 31;
    const int wid   = tid >> 5;
    const int l16   = lane & 15;
    const int hi8   = (lane & 16) ? 8 : 0;
    const int qbase = qtile * 64 + wid * 16;
    const float SCALE = 0.125f;              // 1/sqrt(64)

    // Q fragments (B operand of S^T gemm): rows = queries, 64 dk in 2 frags.
    const _Float16* qp = qh + ((size_t)(b * S + qbase)) * D + h * DK;
    const v16h qf0 = frag16(qp,      D, lane);
    const v16h qf1 = frag16(qp + 32, D, lane);

    v8f zero = {};
    v8f accO[4];
#pragma unroll
    for (int t = 0; t < 4; ++t) accO[t] = zero;
    float mrun = -1e30f;
    float lrun = 0.0f;

    const _Float16* kbh = kh + ((size_t)(b * S)) * D + h * DK;
    const _Float16* vbh = vT + ((size_t)bh) * DK * S;

    for (int kb = 0; kb < S; kb += 32) {
        // --- scores S^T for 32 keys (two 16-key tiles) ---
        v8f sa = zero, sb = zero;
        {
            const _Float16* ka = kbh + (size_t)kb * D;
            sa = wmma_f16(frag16(ka,      D, lane), qf0, sa);
            sa = wmma_f16(frag16(ka + 32, D, lane), qf1, sa);
            const _Float16* kc = kbh + (size_t)(kb + 16) * D;
            sb = wmma_f16(frag16(kc,      D, lane), qf0, sb);
            sb = wmma_f16(frag16(kc + 32, D, lane), qf1, sb);
        }

        // --- online softmax (per query = lane&15; partner lane = lane^16) ---
        float bm = -1e30f;
#pragma unroll
        for (int i = 0; i < 8; ++i) {
            sa[i] *= SCALE; sb[i] *= SCALE;
            bm = fmaxf(bm, fmaxf(sa[i], sb[i]));
        }
        bm = fmaxf(bm, __shfl_xor(bm, 16, 32));
        const float mnew  = fmaxf(mrun, bm);
        const float alpha = __expf(mrun - mnew);
        float p0[8], p1[8], psum = 0.0f;
#pragma unroll
        for (int i = 0; i < 8; ++i) {
            p0[i] = __expf(sa[i] - mnew);
            p1[i] = __expf(sb[i] - mnew);
            psum += p0[i] + p1[i];
        }
        lrun = lrun * alpha + psum;
        mrun = mnew;

        // Rescale O accumulators: per-row alpha broadcast via shfl.
#pragma unroll
        for (int i = 0; i < 8; ++i) {
            const float ai = __shfl(alpha, i + hi8, 32);
#pragma unroll
            for (int t = 0; t < 4; ++t) accO[t][i] *= ai;
        }

        // Pack P into A-fragment layout: elements 0..7 <- sa, 8..15 <- sb.
        v16h pf;
#pragma unroll
        for (int i = 0; i < 8; ++i) {
            pf[i]     = (_Float16)p0[i];
            pf[i + 8] = (_Float16)p1[i];
        }

        // --- O += P * V over 32 keys, 4 dk-tiles of 16 ---
#pragma unroll
        for (int t = 0; t < 4; ++t) {
            const v16h vf = frag16(vbh + (size_t)(t * 16) * S + kb, S, lane);
            accO[t] = wmma_f16(pf, vf, accO[t]);
        }
    }

    // Final 1/l scaling and store (row = query, col = dk).
    const float ltot = lrun + __shfl_xor(lrun, 16, 32);
    const float linv = 1.0f / ltot;
#pragma unroll
    for (int i = 0; i < 8; ++i) {
        const float li = __shfl(linv, i + hi8, 32);
        const int q = qbase + i + hi8;
        _Float16* op = oh + ((size_t)(b * S + q)) * D + h * DK + l16;
#pragma unroll
        for (int t = 0; t < 4; ++t)
            op[t * 16] = (_Float16)(accO[t][i] * li);
    }
}

// ---------------------------------------------------------------------------
// Fused residual + LayerNorm. One block per token row (D=1024, 256 threads).
// ---------------------------------------------------------------------------
__global__ __launch_bounds__(256)
void layernorm_kernel(const float* __restrict__ a, const float* __restrict__ r,
                      const float* __restrict__ g, const float* __restrict__ be,
                      float* __restrict__ o32, _Float16* __restrict__ o16) {
    const int row  = blockIdx.x;
    const int tid  = threadIdx.x;
    const int lane = tid & 31;
    const int wid  = tid >> 5;
    __shared__ float red[8];

    float v[4];
    float s = 0.0f;
#pragma unroll
    for (int j = 0; j < 4; ++j) {
        const int c = tid + j * 256;
        v[j] = a[(size_t)row * D + c] + r[(size_t)row * D + c];
        s += v[j];
    }
#pragma unroll
    for (int off = 16; off; off >>= 1) s += __shfl_xor(s, off, 32);
    if (lane == 0) red[wid] = s;
    __syncthreads();
    float tot = 0.0f;
#pragma unroll
    for (int w = 0; w < 8; ++w) tot += red[w];
    const float mean = tot * (1.0f / D);
    __syncthreads();

    float vs = 0.0f;
#pragma unroll
    for (int j = 0; j < 4; ++j) {
        const float d = v[j] - mean;
        vs += d * d;
    }
#pragma unroll
    for (int off = 16; off; off >>= 1) vs += __shfl_xor(vs, off, 32);
    if (lane == 0) red[wid] = vs;
    __syncthreads();
    float vtot = 0.0f;
#pragma unroll
    for (int w = 0; w < 8; ++w) vtot += red[w];
    const float rstd = rsqrtf(vtot * (1.0f / D) + EPS);

#pragma unroll
    for (int j = 0; j < 4; ++j) {
        const int c = tid + j * 256;
        const float o = g[c] * (v[j] - mean) * rstd + be[c];
        o32[(size_t)row * D + c] = o;
        if (o16) o16[(size_t)row * D + c] = (_Float16)o;
    }
}

// ---------------------------------------------------------------------------
// Host orchestration
// ---------------------------------------------------------------------------
extern "C" void kernel_launch(void* const* d_in, const int* in_sizes, int n_in,
                              void* d_out, int out_size, void* d_ws, size_t ws_size,
                              hipStream_t stream) {
    const float* x  = (const float*)d_in[0];
    const float* Wq = (const float*)d_in[1];
    const float* bq = (const float*)d_in[2];
    const float* Wk = (const float*)d_in[3];
    const float* bk = (const float*)d_in[4];
    const float* Wv = (const float*)d_in[5];
    const float* bv = (const float*)d_in[6];
    const float* Wo = (const float*)d_in[7];
    const float* bo = (const float*)d_in[8];
    const float* W1 = (const float*)d_in[9];
    const float* b1 = (const float*)d_in[10];
    const float* W2 = (const float*)d_in[11];
    const float* b2 = (const float*)d_in[12];
    const float* g1 = (const float*)d_in[13];
    const float* be1= (const float*)d_in[14];
    const float* g2 = (const float*)d_in[15];
    const float* be2= (const float*)d_in[16];
    float* out = (float*)d_out;

    // ---- workspace layout ----
    char* wsb = (char*)d_ws;
    size_t off = 0;
    auto take = [&](size_t bytes) {
        char* p = wsb + off;
        off = (off + bytes + 255) & ~(size_t)255;
        return p;
    };
    _Float16* xh    = (_Float16*)take((size_t)M * D * 2);
    _Float16* WqT   = (_Float16*)take((size_t)D * D * 2);
    _Float16* WkT   = (_Float16*)take((size_t)D * D * 2);
    _Float16* WvT   = (_Float16*)take((size_t)D * D * 2);
    _Float16* WoT   = (_Float16*)take((size_t)D * D * 2);
    _Float16* W1T   = (_Float16*)take((size_t)DFF * D * 2);
    _Float16* W2T   = (_Float16*)take((size_t)D * DFF * 2);
    _Float16* qhb   = (_Float16*)take((size_t)M * D * 2);
    _Float16* khb   = (_Float16*)take((size_t)M * D * 2);
    _Float16* vhb   = (_Float16*)take((size_t)M * D * 2);
    _Float16* vTh   = (_Float16*)take((size_t)M * D * 2);
    _Float16* attnh = (_Float16*)take((size_t)M * D * 2);
    _Float16* h1h   = (_Float16*)take((size_t)M * D * 2);
    _Float16* ffn1h = (_Float16*)take((size_t)M * DFF * 2);
    float*    oproj = (float*)take((size_t)M * D * 4);
    float*    h1    = (float*)take((size_t)M * D * 4);
    float*    ffn2  = (float*)take((size_t)M * D * 4);
    (void)ws_size; (void)n_in; (void)in_sizes; (void)out_size;

    // ---- phase 0: precision / layout conversion ----
    cvt_f32_to_f16<<<(M * D) / 256, 256, 0, stream>>>(x, xh, (long)M * D);
    transpose_cvt<<<(D * D) / 256, 256, 0, stream>>>(Wq, WqT, D, D);
    transpose_cvt<<<(D * D) / 256, 256, 0, stream>>>(Wk, WkT, D, D);
    transpose_cvt<<<(D * D) / 256, 256, 0, stream>>>(Wv, WvT, D, D);
    transpose_cvt<<<(D * D) / 256, 256, 0, stream>>>(Wo, WoT, D, D);
    transpose_cvt<<<(D * DFF) / 256, 256, 0, stream>>>(W1, W1T, D, DFF);
    transpose_cvt<<<(DFF * D) / 256, 256, 0, stream>>>(W2, W2T, DFF, D);

    const dim3 blk256(256), blk128(128);
    const dim3 gD(D / 128, M / 128);        // N=1024 GEMMs
    const dim3 gF(DFF / 128, M / 128);      // N=4096 GEMM

    // ---- QKV projections (f16 out) ----
    gemm_wmma<_Float16, false><<<gD, blk256, 0, stream>>>(xh, WqT, bq, qhb, M, D, D);
    gemm_wmma<_Float16, false><<<gD, blk256, 0, stream>>>(xh, WkT, bk, khb, M, D, D);
    gemm_wmma<_Float16, false><<<gD, blk256, 0, stream>>>(xh, WvT, bv, vhb, M, D, D);
    vtrans_kernel<<<(M * D) / 256, 256, 0, stream>>>(vhb, vTh);

    // ---- attention ----
    attn_wmma<<<dim3(S / 64, B * H), blk128, 0, stream>>>(qhb, khb, vTh, attnh);

    // ---- output projection + LN1 ----
    gemm_wmma<float, false><<<gD, blk256, 0, stream>>>(attnh, WoT, bo, oproj, M, D, D);
    layernorm_kernel<<<M, blk256, 0, stream>>>(oproj, x, g1, be1, h1, h1h);

    // ---- FFN + LN2 ----
    gemm_wmma<_Float16, true><<<gF, blk256, 0, stream>>>(h1h, W1T, b1, ffn1h, M, DFF, D);
    gemm_wmma<float, false><<<gD, blk256, 0, stream>>>(ffn1h, W2T, b2, ffn2, M, D, DFF);
    layernorm_kernel<<<M, blk256, 0, stream>>>(ffn2, h1, g2, be2, out, (_Float16*)nullptr);
}
